// CausalAttention_82678120448079
// MI455X (gfx1250) — matrix operand order repaired and checked by reading, not compiled, not run on hardware
//
#include <hip/hip_runtime.h>

// ---------------------------------------------------------------------------
// Types
// ---------------------------------------------------------------------------
typedef _Float16 v16h __attribute__((ext_vector_type(16)));
typedef _Float16 h8   __attribute__((ext_vector_type(8)));
typedef float    v8f  __attribute__((ext_vector_type(8)));
typedef float    f32x8 __attribute__((ext_vector_type(8)));

static __device__ __forceinline__ v16h mk16(h8 lo, h8 hi) {
  v16h r;
#pragma unroll
  for (int i = 0; i < 8; ++i) { r[i] = lo[i]; r[i + 8] = hi[i]; }
  return r;
}

// B fragment (32x16 f16): per lane, 16 contiguous elems at p
static __device__ __forceinline__ v16h loadB(const _Float16* p) {
  return mk16(*(const h8*)p, *(const h8*)(p + 8));
}

static __device__ __forceinline__ v8f wmma16(v16h a, v16h b, v8f c) {
  return __builtin_amdgcn_wmma_f32_16x16x32_f16(false, a, false, b,
                                                (short)0, c, false, false);
}

// gfx1250 async memory->LDS copy (16B per lane), tracked by ASYNCcnt.
// Low 32 bits of a generic pointer into LDS are the LDS byte offset
// (flat aperture mapping).
static __device__ __forceinline__ void async_ld_b128(void* lds, const void* gaddr) {
  asm volatile("global_load_async_to_lds_b128 %0, %1, off"
               :
               : "v"((unsigned)(unsigned long long)lds),
                 "v"((unsigned long long)gaddr)
               : "memory");
}
#define WAIT_ASYNC(n) asm volatile("s_wait_asynccnt " #n ::: "memory")

// ---------------------------------------------------------------------------
// Constants
// ---------------------------------------------------------------------------
#define BSZ 8
#define SEQ 1024
#define NHD 8
#define DH 64
#define CHN 512

// ---------------------------------------------------------------------------
// Kernel 1: weight normalization -> transposed f16 weights
// wT[ch][c] = f16( v[c][ch] * g[ch] / ||v[:,ch]|| )
// ---------------------------------------------------------------------------
__global__ __launch_bounds__(256) void wn_kernel(const float* __restrict__ v,
                                                 const float* __restrict__ g,
                                                 _Float16* __restrict__ wT,
                                                 int Cin) {
  __shared__ float red[256];
  const int ch = blockIdx.x;
  const int tid = threadIdx.x;
  float acc = 0.f;
  for (int c = tid; c < Cin; c += 256) {
    float t = v[(size_t)c * CHN + ch];
    acc += t * t;
  }
  red[tid] = acc;
  __syncthreads();
  for (int s = 128; s > 0; s >>= 1) {
    if (tid < s) red[tid] += red[tid + s];
    __syncthreads();
  }
  const float scale = g[ch] / sqrtf(red[0]);
  for (int c = tid; c < Cin; c += 256) {
    wT[(size_t)ch * Cin + c] = (_Float16)(v[(size_t)c * CHN + ch] * scale);
  }
}

// ---------------------------------------------------------------------------
// Kernel 2: projection GEMM
//   vtrans==0: out[b,h,s,d] = f16((X @ W + bias) * scale)       (Q, K)
//   vtrans==1: out[b,h,d,s] = f16((X @ W + bias) * scale)               (V, transposed)
// grid.x = (B*S)/128 row tiles, grid.y = CH/64 channel blocks (== head h)
// ---------------------------------------------------------------------------
__global__ __launch_bounds__(256) void proj_kernel(const float* __restrict__ X,
                                                   const _Float16* __restrict__ wT,
                                                   const float* __restrict__ bias,
                                                   _Float16* __restrict__ out,
                                                   int Cin, float scale, int vtrans) {
  const int tid = threadIdx.x;
  const int w = tid >> 5;
  const int lane = tid & 31;
  const int ln = lane & 15;
  const bool hi = lane >= 16;
  const int r0 = blockIdx.x * 128 + w * 16;
  const int chb = blockIdx.y * 64;
  const int h = blockIdx.y;

  const float* xrow = X + (size_t)(r0 + ln) * Cin + (hi ? 16 : 0);

  v8f acc[4] = {};
  for (int kb = 0; kb < Cin; kb += 32) {
    f32x8 x0 = *(const f32x8*)(xrow + kb);
    f32x8 x1 = *(const f32x8*)(xrow + kb + 8);
    v16h a;
#pragma unroll
    for (int i = 0; i < 8; ++i) {
      a[i] = (_Float16)x0[i];
      a[i + 8] = (_Float16)x1[i];
    }
    const _Float16* wp = wT + (size_t)(chb + ln) * Cin + kb + (hi ? 16 : 0);
#pragma unroll
    for (int t = 0; t < 4; ++t) {
      acc[t] = wmma16(a, loadB(wp + (size_t)(16 * t) * Cin), acc[t]);
    }
  }

  if (vtrans) {
    // transposed store: per lane, 8 consecutive s for fixed d -> one b128
    const int R0 = r0 + (hi ? 8 : 0);
    const int b = R0 >> 10;
    const int s0 = R0 & 1023;
#pragma unroll
    for (int t = 0; t < 4; ++t) {
      const int d = t * 16 + ln;
      const float bv = bias[chb + d];
      h8 pk;
#pragma unroll
      for (int v = 0; v < 8; ++v) pk[v] = (_Float16)((acc[t][v] + bv) * scale);
      *(h8*)(out + (((size_t)b * NHD + h) * DH + d) * SEQ + s0) = pk;
    }
  } else {
#pragma unroll
    for (int t = 0; t < 4; ++t) {
      const int d = t * 16 + ln;
      const float bv = bias[chb + d];
#pragma unroll
      for (int v = 0; v < 8; ++v) {
        const int R = r0 + v + (hi ? 8 : 0);
        const int b = R >> 10;
        const int s = R & 1023;
        out[(((size_t)b * NHD + h) * SEQ + s) * DH + d] =
            (_Float16)((acc[t][v] + bv) * scale);
      }
    }
  }
}

// ---------------------------------------------------------------------------
// Kernel 3: causal flash attention, async double-buffered K/V tiles
// grid.x = S/128 query tiles, grid.y = B*NH
// workgroup: 8 waves x 16 query rows; K/V tiles of 32 keys in LDS
// K layout [bh][s][d]; V layout [bh][d][s] (pre-transposed by proj)
// ---------------------------------------------------------------------------
#define KSTR 72   // K LDS row stride (f16): 144B, 16B aligned, conflict-free
#define VSTR 40   // V^T LDS row stride (f16): 80B
#define PSTR 40   // P LDS row stride (f16): 80B
#define KTILE (32 * KSTR)
#define VTILE (64 * VSTR)

__global__ __launch_bounds__(256) void attn_kernel(const _Float16* __restrict__ Qh,
                                                   const _Float16* __restrict__ Kh,
                                                   const _Float16* __restrict__ Vh,
                                                   float* __restrict__ out) {
  __shared__ alignas(16) _Float16 Kl[2 * KTILE];
  __shared__ alignas(16) _Float16 Vt[2 * VTILE];
  __shared__ alignas(16) _Float16 Pl[8 * 16 * PSTR];

  const int tid = threadIdx.x;
  const int w = tid >> 5;
  const int lane = tid & 31;
  const int ln = lane & 15;
  const bool hi = lane >= 16;
  const int qt = blockIdx.x;
  const int bh = blockIdx.y;
  const int qb = qt * 128;
  const int qw = qb + w * 16;

  // Q A-fragments for this wave's 16 rows (loop invariant); 1/8 already folded
  const _Float16* qrow = Qh + ((size_t)bh * SEQ + qw + ln) * DH + (hi ? 16 : 0);
  const v16h qa0 = mk16(*(const h8*)qrow, *(const h8*)(qrow + 8));
  const v16h qa1 = mk16(*(const h8*)(qrow + 32), *(const h8*)(qrow + 40));

  v8f o0 = {}, o1 = {}, o2 = {}, o3 = {};
  float mr[8], lr[8];
#pragma unroll
  for (int v = 0; v < 8; ++v) { mr[v] = -1.0e30f; lr[v] = 0.f; }

  const int nkt = qb / 32 + 4;  // keys [0, qb+128) cover all k < q in this tile
  const _Float16* Kg = Kh + (size_t)bh * SEQ * DH;
  const _Float16* Vg = Vh + (size_t)bh * DH * SEQ;
  _Float16* pw = Pl + w * 16 * PSTR;

  // per-thread async-copy coordinates (16B per lane, 4KB per tile each)
  const int kr = tid >> 3, kc = (tid & 7) << 3;   // K: row s, col d
  const int vr = tid >> 2, vc = (tid & 3) << 3;   // V: row d, col s

  // prologue: prefetch tile 0 into buffer 0
  async_ld_b128(Kl + kr * KSTR + kc, Kg + (size_t)kr * DH + kc);
  async_ld_b128(Vt + vr * VSTR + vc, Vg + (size_t)vr * SEQ + vc);

  for (int it = 0; it < nkt; ++it) {
    const int kb = it * 32;
    const int buf = it & 1;
    __syncthreads();  // all waves done reading buffer buf^1 (prev compute)
    if (it + 1 < nkt) {
      const int nb = it * 32 + 32;
      async_ld_b128(Kl + (buf ^ 1) * KTILE + kr * KSTR + kc,
                    Kg + (size_t)(nb + kr) * DH + kc);
      async_ld_b128(Vt + (buf ^ 1) * VTILE + vr * VSTR + vc,
                    Vg + (size_t)vr * SEQ + nb + vc);
      WAIT_ASYNC(0x2);  // tile it complete; tile it+1 still in flight
    } else {
      WAIT_ASYNC(0x0);
    }
    __syncthreads();  // tile it visible to all waves

    if (kb > qw + 14) continue;  // wave-uniform: fully masked for this wave

    // scores: S[16q x 32k] = Q(16x64) @ K^T, two 16x16 tiles
    const _Float16* kp0 = Kl + buf * KTILE + ln * KSTR + (hi ? 16 : 0);
    const _Float16* kp1 = Kl + buf * KTILE + (ln + 16) * KSTR + (hi ? 16 : 0);
    v8f s0 = {}, s1 = {};
    s0 = wmma16(qa0, loadB(kp0), s0);
    s0 = wmma16(qa1, loadB(kp0 + 32), s0);
    s1 = wmma16(qa0, loadB(kp1), s1);
    s1 = wmma16(qa1, loadB(kp1 + 32), s1);

    // causal mask (strictly lower triangular): key k allowed iff k < q
    if (kb + 31 >= qw) {
#pragma unroll
      for (int v = 0; v < 8; ++v) {
        const int qg = qw + v + (hi ? 8 : 0);
        s0[v] = (kb + ln < qg) ? s0[v] : -10000.0f;
        s1[v] = (kb + 16 + ln < qg) ? s1[v] : -10000.0f;
      }
    }

    // online softmax: row reductions across the 16 lanes of each half-wave
    float mt[8];
#pragma unroll
    for (int v = 0; v < 8; ++v) mt[v] = fmaxf(s0[v], s1[v]);
#pragma unroll
    for (int mk = 1; mk <= 8; mk <<= 1) {
#pragma unroll
      for (int v = 0; v < 8; ++v)
        mt[v] = fmaxf(mt[v], __shfl_xor(mt[v], mk, 32));
    }
    float p0[8], p1[8], rs[8], al[8];
#pragma unroll
    for (int v = 0; v < 8; ++v) {
      const float mn = fmaxf(mr[v], mt[v]);
      al[v] = __expf(mr[v] - mn);
      mr[v] = mn;
      p0[v] = __expf(s0[v] - mn);
      p1[v] = __expf(s1[v] - mn);
      rs[v] = p0[v] + p1[v];
    }
#pragma unroll
    for (int mk = 1; mk <= 8; mk <<= 1) {
#pragma unroll
      for (int v = 0; v < 8; ++v) rs[v] += __shfl_xor(rs[v], mk, 32);
    }
    v8f alv;
#pragma unroll
    for (int v = 0; v < 8; ++v) {
      lr[v] = lr[v] * al[v] + rs[v];
      alv[v] = al[v];
    }
    o0 *= alv; o1 *= alv; o2 *= alv; o3 *= alv;

    // P (C layout) -> LDS -> A-layout fragment (per-wave region, HW in-order)
#pragma unroll
    for (int v = 0; v < 8; ++v) {
      const int m = v + (hi ? 8 : 0);
      pw[m * PSTR + ln] = (_Float16)p0[v];
      pw[m * PSTR + 16 + ln] = (_Float16)p1[v];
    }
    const _Float16* pap = pw + ln * PSTR + (hi ? 16 : 0);
    const v16h pa = mk16(*(const h8*)pap, *(const h8*)(pap + 8));

    // O += P(16x32) @ V(32x64); V tile rows are ready-made B fragments
    const _Float16* vb = Vt + buf * VTILE + (hi ? 16 : 0);
    o0 = wmma16(pa, loadB(vb + (0 * 16 + ln) * VSTR), o0);
    o1 = wmma16(pa, loadB(vb + (1 * 16 + ln) * VSTR), o1);
    o2 = wmma16(pa, loadB(vb + (2 * 16 + ln) * VSTR), o2);
    o3 = wmma16(pa, loadB(vb + (3 * 16 + ln) * VSTR), o3);
  }

  // normalize + store fp32 output [B, S, CH]; zero row q==0 (start mask)
  const int b = bh >> 3;
  const int hh = bh & 7;
  float* ob = out + (size_t)b * SEQ * CHN + (size_t)hh * DH;
#pragma unroll
  for (int v = 0; v < 8; ++v) {
    const int m = v + (hi ? 8 : 0);
    const int qg = qw + m;
    const float inv = 1.0f / lr[v];
    float* orow = ob + (size_t)qg * CHN;
    orow[0 * 16 + ln] = (qg == 0) ? 0.f : o0[v] * inv;
    orow[1 * 16 + ln] = (qg == 0) ? 0.f : o1[v] * inv;
    orow[2 * 16 + ln] = (qg == 0) ? 0.f : o2[v] * inv;
    orow[3 * 16 + ln] = (qg == 0) ? 0.f : o3[v] * inv;
  }
}

// ---------------------------------------------------------------------------
// Launch
// ---------------------------------------------------------------------------
extern "C" void kernel_launch(void* const* d_in, const int* in_sizes, int n_in,
                              void* d_out, int out_size, void* d_ws, size_t ws_size,
                              hipStream_t stream) {
  (void)in_sizes; (void)n_in; (void)out_size; (void)ws_size;
  const float* query = (const float*)d_in[0];
  const float* key   = (const float*)d_in[1];
  const float* vq = (const float*)d_in[2];
  const float* gq = (const float*)d_in[3];
  const float* bq = (const float*)d_in[4];
  const float* vk = (const float*)d_in[5];
  const float* gk = (const float*)d_in[6];
  const float* bk = (const float*)d_in[7];
  const float* vv = (const float*)d_in[8];
  const float* gv = (const float*)d_in[9];
  const float* bv = (const float*)d_in[10];
  float* out = (float*)d_out;

  char* ws = (char*)d_ws;
  size_t off = 0;
  _Float16* wqT = (_Float16*)(ws + off); off += (size_t)CHN * 256 * 2;   // 256 KB
  _Float16* wkT = (_Float16*)(ws + off); off += (size_t)CHN * 512 * 2;   // 512 KB
  _Float16* wvT = (_Float16*)(ws + off); off += (size_t)CHN * 512 * 2;   // 512 KB
  const size_t qkv = (size_t)BSZ * NHD * SEQ * DH * 2;                   // 8 MB each
  _Float16* Qh = (_Float16*)(ws + off); off += qkv;
  _Float16* Kh = (_Float16*)(ws + off); off += qkv;
  _Float16* Vh = (_Float16*)(ws + off); off += qkv;   // [b,h,d,s] transposed

  // 1) weight normalization -> transposed f16 weights
  wn_kernel<<<dim3(CHN), dim3(256), 0, stream>>>(vq, gq, wqT, 256);
  wn_kernel<<<dim3(CHN), dim3(256), 0, stream>>>(vk, gk, wkT, 512);
  wn_kernel<<<dim3(CHN), dim3(256), 0, stream>>>(vv, gv, wvT, 512);

  // 2) projections (fold 1/sqrt(DH)=0.125 into Q; V stored transposed)
  const dim3 pgrid(64, 8);
  proj_kernel<<<pgrid, dim3(256), 0, stream>>>(query, wqT, bq, Qh, 256, 0.125f, 0);
  proj_kernel<<<pgrid, dim3(256), 0, stream>>>(key,   wkT, bk, Kh, 512, 1.0f, 0);
  proj_kernel<<<pgrid, dim3(256), 0, stream>>>(key,   wvT, bv, Vh, 512, 1.0f, 1);

  // 3) causal flash attention (async double-buffered K/V tiles)
  attn_kernel<<<dim3(SEQ / 128, BSZ * NHD), dim3(256), 0, stream>>>(Qh, Kh, Vh, out);
}
